// Plan_GRU_1254130450916
// MI455X (gfx1250) — matrix operand order repaired and checked by reading, not compiled, not run on hardware
//
#include <hip/hip_runtime.h>
#include <hip/hip_bf16.h>

// ---------------------------------------------------------------------------
// Plan_GRU for MI455X (gfx1250): bf16 WMMA (v_wmma_f32_16x16x32_bf16) with
// fp32 accumulation.  Teacher forcing => input-side gates for all 30 steps
// are one big parallel GEMM; only gh = h @ W_hh^T and the attention
// projection are serial.  Weights stay in native [N,K] row-major layout so
// WMMA B-fragments (lane n needs K-contiguous data) are two ds_load_b128s.
// Global->LDS staging uses gfx1250 async copies
// (global_load_async_to_lds_b128 + s_wait_asynccnt).
// ---------------------------------------------------------------------------

typedef __attribute__((ext_vector_type(16))) __bf16 v16bf;
typedef __attribute__((ext_vector_type(8)))  float  v8f;

constexpr int B  = 512;
constexpr int L  = 30;
constexpr int H  = 1024;
constexpr int T  = 30;
constexpr int H3 = 3 * H;

// ---- async global->LDS copy (CDNA5 ISA op 98) ----------
// LDS byte address = low 32 bits of the flat pointer (ISA §10.2 aperture).
__device__ inline void async_copy_b128(const void* gptr, void* lptr) {
  unsigned            lds = (unsigned)(unsigned long long)lptr;
  unsigned long long  g   = (unsigned long long)gptr;
  asm volatile("global_load_async_to_lds_b128 %0, %1, off"
               :: "v"(lds), "v"(g) : "memory");
}
__device__ inline void wait_async() {
  asm volatile("s_wait_asynccnt 0" ::: "memory");
}

// ---- WMMA fragment helpers (contiguous K order k = 16*(lane/16)+s, same for A and B) ------
// A (16x32 MxK): lane<16 -> M=lane, K {0..15}; lane>=16 -> K {16..31}
// Two aligned 16B runs per lane.
__device__ inline v16bf frag_a(const __bf16* p, int ld) {
  const int lane = threadIdx.x & 31;
  const __bf16* q = p + (size_t)(lane & 15) * ld + ((lane >> 4) << 4);
  union { uint4 u[2]; v16bf v; } t;
  t.u[0] = *(const uint4*)q;         // K = kh .. kh+7
  t.u[1] = *(const uint4*)(q + 8);   // K = kh+8 .. kh+15
  return t.v;
}

// B (32x16 KxN) staged N-major: lane col picks row n, lane half picks K 16-run.
// One contiguous 32B run per lane.
__device__ inline v16bf frag_b(const __bf16* p, int ld) {
  const int lane = threadIdx.x & 31;
  const __bf16* q = p + (size_t)(lane & 15) * ld + ((lane >> 4) << 4);
  union { uint4 u[2]; v16bf v; } t;
  t.u[0] = *(const uint4*)q;
  t.u[1] = *(const uint4*)(q + 8);
  return t.v;
}

__device__ inline v8f wmma_bf16(v16bf a, v16bf b, v8f c) {
  return __builtin_amdgcn_wmma_f32_16x16x32_bf16(false, a, false, b,
                                                 (short)0, c, false, false);
}

__device__ inline float sigm(float x) { return 1.f / (1.f + __expf(-x)); }

// ---------------------------------------------------------------------------
// Prep kernels
// ---------------------------------------------------------------------------
__global__ void cast_f32_to_bf16(const float* __restrict__ src,
                                 __bf16* __restrict__ dst, int n) {
  int i = blockIdx.x * 256 + threadIdx.x;
  if (i < n) dst[i] = (__bf16)src[i];
}

// gi0[n] = b_ih[n] + S . W_ih[n,:]   (one row, shared by all batches at t=0)
__global__ void gi0_kernel(const float* __restrict__ S,
                           const float* __restrict__ W_ih,
                           const float* __restrict__ b_ih,
                           float* __restrict__ gi0) {
  int n = blockIdx.x * 256 + threadIdx.x;
  if (n < H3) {
    float s = b_ih[n];
    const float* wr = W_ih + (size_t)n * H;
    for (int k = 0; k < H; ++k) s += S[k] * wr[k];
    gi0[n] = s;
  }
}

// h0 = mean_L(input_hidden) -> fp32 + bf16 copies
__global__ void mean_kernel(const float* __restrict__ ih,
                            float* __restrict__ h0,
                            __bf16* __restrict__ h0_bf) {
  int i = blockIdx.x * 256 + threadIdx.x;
  if (i < B * H) {
    int b = i / H, h = i - b * H;
    float s = 0.f;
    const float* p = ih + ((size_t)b * L) * H + h;
    for (int l = 0; l < L; ++l) s += p[(size_t)l * H];
    s *= (1.0f / L);
    h0[i]    = s;
    h0_bf[i] = (__bf16)s;
  }
}

// ---------------------------------------------------------------------------
// Big parallel GEMM:  C[M,N] = A[M,K](bf16) @ W[N,K]^T(bf16) + bias[N] (fp32)
// block tile 64x128, 8 waves, each wave 32x32 (2x2 wmma tiles), BK=32
// ---------------------------------------------------------------------------
__global__ __launch_bounds__(256) void gemm_bias_kernel(
    const __bf16* __restrict__ A, const __bf16* __restrict__ W,
    const float* __restrict__ bias, float* __restrict__ C,
    int M, int N, int K) {
  __shared__ __bf16 As[64][40];     // 80B row stride (16B aligned)
  __shared__ __bf16 Bs[128][40];    // N-major, same layout as A

  const int tid  = threadIdx.x;
  const int wid  = tid >> 5;
  const int lane = tid & 31;
  const int bm = blockIdx.x * 64;
  const int bn = blockIdx.y * 128;
  const int wm = (wid >> 2) * 32;   // 0,32
  const int wn = (wid & 3) * 32;    // 0,32,64,96

  v8f acc[2][2];
#pragma unroll
  for (int i = 0; i < 2; ++i)
#pragma unroll
    for (int j = 0; j < 2; ++j) acc[i][j] = (v8f){0,0,0,0,0,0,0,0};

  for (int k0 = 0; k0 < K; k0 += 32) {
    {  // A: 64x32 = 256 x b128, one per thread (async -> LDS)
      int r = tid >> 2, cv = tid & 3;
      async_copy_b128(A + (size_t)(bm + r) * K + k0 + cv * 8,
                      &As[r][cv * 8]);
    }
#pragma unroll
    for (int rep = 0; rep < 2; ++rep) {  // B: 128 n-rows x 32 k = 512 x b128
      int u = tid + rep * 256;
      int r = u >> 2, cv = u & 3;
      async_copy_b128(W + (size_t)(bn + r) * K + k0 + cv * 8,
                      &Bs[r][cv * 8]);
    }
    wait_async();
    __syncthreads();
    if (k0 + 32 < K) {  // gfx1250 global_prefetch_b8 for next K slab
      __builtin_prefetch(A + (size_t)(bm + (tid >> 2)) * K + k0 + 32, 0, 1);
      __builtin_prefetch(W + (size_t)(bn + (tid >> 2)) * K + k0 + 32, 0, 1);
    }
    v16bf af[2], bfr[2];
    af[0]  = frag_a(&As[wm][0], 40);
    af[1]  = frag_a(&As[wm + 16][0], 40);
    bfr[0] = frag_b(&Bs[wn][0], 40);
    bfr[1] = frag_b(&Bs[wn + 16][0], 40);
#pragma unroll
    for (int i = 0; i < 2; ++i)
#pragma unroll
      for (int j = 0; j < 2; ++j)
        acc[i][j] = wmma_bf16(af[i], bfr[j], acc[i][j]);
    __syncthreads();
  }

  const int nloc = lane & 15, mh = (lane >> 4) * 8;
#pragma unroll
  for (int i = 0; i < 2; ++i)
#pragma unroll
    for (int j = 0; j < 2; ++j) {
      int cm = bm + wm + i * 16 + mh;
      int cn = bn + wn + j * 16 + nloc;
      float bv = bias[cn];
#pragma unroll
      for (int v = 0; v < 8; ++v)
        C[(size_t)(cm + v) * N + cn] = acc[i][j][v] + bv;
    }
}

// ---------------------------------------------------------------------------
// Serial step 1: h' = GRU(h, gi_t)   -- fused gh GEMM (3 gates) + gate math
// grid (B/64, H/64), 8 waves, each wave: 16(M) x 32(N) x 3 gates = 6 accs
// ---------------------------------------------------------------------------
__global__ __launch_bounds__(256) void step_h_kernel(
    const float* __restrict__ h_cur, const __bf16* __restrict__ h_cur_bf,
    float* __restrict__ h_nxt, __bf16* __restrict__ h_nxt_bf,
    const __bf16* __restrict__ Whh,     // [3H, H] native row-major
    const float* __restrict__ b_hh,
    const float* __restrict__ GI,       // [B*T, 3H] (row b*T+t)
    const float* __restrict__ gi0, int giRow, int use_gi0) {
  __shared__ __bf16 As[64][40];
  __shared__ __bf16 Bs[3][64][40];      // per-gate, N-major

  const int tid  = threadIdx.x;
  const int wid  = tid >> 5;
  const int lane = tid & 31;
  const int b0 = blockIdx.x * 64;
  const int bn = blockIdx.y * 64;
  const int wm = (wid >> 1) * 16;       // 0,16,32,48
  const int wn = (wid & 1) * 32;        // 0,32

  v8f acc[3][2];
#pragma unroll
  for (int g = 0; g < 3; ++g)
#pragma unroll
    for (int j = 0; j < 2; ++j) acc[g][j] = (v8f){0,0,0,0,0,0,0,0};

  for (int k0 = 0; k0 < H; k0 += 32) {
    {  // A: h tile 64x32
      int r = tid >> 2, cv = tid & 3;
      async_copy_b128(h_cur_bf + (size_t)(b0 + r) * H + k0 + cv * 8,
                      &As[r][cv * 8]);
    }
#pragma unroll
    for (int rep = 0; rep < 3; ++rep) {  // B: 3 gate tiles, 64 n-rows x 32 k
      int u = tid + rep * 256;
      int g = u >> 8, rem = u & 255;
      int r = rem >> 2, cv = rem & 3;
      async_copy_b128(Whh + (size_t)(g * H + bn + r) * H + k0 + cv * 8,
                      &Bs[g][r][cv * 8]);
    }
    wait_async();
    __syncthreads();
    if (k0 + 32 < H)
      __builtin_prefetch(h_cur_bf + (size_t)(b0 + (tid >> 2)) * H + k0 + 32,
                         0, 1);
    v16bf a = frag_a(&As[wm][0], 40);
#pragma unroll
    for (int g = 0; g < 3; ++g)
#pragma unroll
      for (int j = 0; j < 2; ++j)
        acc[g][j] = wmma_bf16(a, frag_b(&Bs[g][wn + j * 16][0], 40),
                              acc[g][j]);
    __syncthreads();
  }

  const int nloc = lane & 15, mh = (lane >> 4) * 8;
#pragma unroll
  for (int j = 0; j < 2; ++j) {
    const int gcol = bn + wn + j * 16 + nloc;
    const float bhr = b_hh[gcol], bhz = b_hh[H + gcol], bhn = b_hh[2 * H + gcol];
#pragma unroll
    for (int v = 0; v < 8; ++v) {
      const int bb = b0 + wm + mh + v;
      float ir, iz, inn;
      if (use_gi0) {
        ir = gi0[gcol]; iz = gi0[H + gcol]; inn = gi0[2 * H + gcol];
      } else {
        const float* g = GI + ((size_t)bb * T + giRow) * H3;
        ir = g[gcol]; iz = g[H + gcol]; inn = g[2 * H + gcol];
      }
      const float r = sigm(ir + acc[0][j][v] + bhr);
      const float z = sigm(iz + acc[1][j][v] + bhz);
      const float n = tanhf(inn + r * (acc[2][j][v] + bhn));
      const float ho = h_cur[(size_t)bb * H + gcol];
      const float hv = (1.f - z) * n + z * ho;
      h_nxt[(size_t)bb * H + gcol]    = hv;
      h_nxt_bf[(size_t)bb * H + gcol] = (__bf16)hv;
    }
  }
}

// ---------------------------------------------------------------------------
// Serial step 2: p = h' @ Wl_w^T + Wl_b (WMMA, p tile kept in LDS as bf16),
// then pre[b,l] = p[b] . input_hidden[b,l]  (per-batch GEMV, VALU).
// grid B/16 blocks, 8 waves; wave w computes 128 columns of p.
// B-fragments come straight from global Wl (row-major => contiguous 32B).
// ---------------------------------------------------------------------------
__global__ __launch_bounds__(256) void step_attn_kernel(
    const __bf16* __restrict__ h_bf, const __bf16* __restrict__ Wl, // [H,H]
    const float* __restrict__ Wl_b, const __bf16* __restrict__ ih_bf,
    float* __restrict__ out, int t) {
  __shared__ __bf16 As[16][40];
  __shared__ __bf16 pL[16][1032];

  const int tid  = threadIdx.x;
  const int wid  = tid >> 5;
  const int lane = tid & 31;
  const int b0 = blockIdx.x * 16;
  const int n0 = wid * 128;

  v8f acc[8];
#pragma unroll
  for (int j = 0; j < 8; ++j) acc[j] = (v8f){0,0,0,0,0,0,0,0};

  for (int k0 = 0; k0 < H; k0 += 32) {
    if (tid < 64) {  // A: 16x32 h' tile (waves 0-1 issue, all waves wait)
      int r = tid >> 2, cv = tid & 3;
      async_copy_b128(h_bf + (size_t)(b0 + r) * H + k0 + cv * 8,
                      &As[r][cv * 8]);
    }
    wait_async();
    __syncthreads();
    v16bf a = frag_a(&As[0][0], 40);
#pragma unroll
    for (int j = 0; j < 8; ++j)
      acc[j] = wmma_bf16(
          a, frag_b(Wl + (size_t)(n0 + j * 16) * H + k0, H), acc[j]);
    __syncthreads();
  }

  const int nloc = lane & 15, mh = (lane >> 4) * 8;
#pragma unroll
  for (int j = 0; j < 8; ++j) {
    const int col = n0 + j * 16 + nloc;
    const float bv = Wl_b[col];
#pragma unroll
    for (int v = 0; v < 8; ++v) pL[mh + v][col] = (__bf16)(acc[j][v] + bv);
  }
  __syncthreads();

  for (int idx = tid; idx < 16 * L; idx += 256) {
    const int m = idx / L, l = idx - m * L;
    const int bb = b0 + m;
    const __bf16* pr = &pL[m][0];
    const __bf16* ir = ih_bf + ((size_t)bb * L + l) * H;
    float s = 0.f;
    for (int k = 0; k < H; k += 8) {
#pragma unroll
      for (int u = 0; u < 8; ++u) s += (float)pr[k + u] * (float)ir[k + u];
    }
    out[((size_t)bb * T + t) * L + l] = s;
  }
}

// ---------------------------------------------------------------------------
// Final: per-batch softmax/argmax/flag chain over T steps.  One wave32 per
// batch row (lane <-> l, L=30, lanes 30-31 idle); shuffle reductions.
// ---------------------------------------------------------------------------
__global__ __launch_bounds__(256) void plan_kernel(
    const int* __restrict__ pos_label, float* __restrict__ out) {
  const int bb   = blockIdx.x * 8 + (threadIdx.x >> 5);
  const int lane = threadIdx.x & 31;
  const size_t INF_OFF = (size_t)B * T * L;

  int cnt = (lane < L && pos_label[(size_t)bb * L + lane] != -1) ? 1 : 0;
  int ss = cnt;
#pragma unroll
  for (int off = 16; off > 0; off >>= 1) ss += __shfl_xor(ss, off, 32);
  const bool valid = lane < ss;

  bool flag = false;
  for (int t = 0; t < T; ++t) {
    const size_t base = ((size_t)bb * T + t) * L;
    float x = (lane < L) ? out[base + lane] : -INFINITY;
    float mval = (flag || !valid || lane >= L) ? -INFINITY : x;

    float mx = mval;
#pragma unroll
    for (int off = 16; off > 0; off >>= 1)
      mx = fmaxf(mx, __shfl_xor(mx, off, 32));
    float e = (mval == -INFINITY) ? 0.f : __expf(mval - mx);
    float sum = e;
#pragma unroll
    for (int off = 16; off > 0; off >>= 1) sum += __shfl_xor(sum, off, 32);
    if (lane < L) out[INF_OFF + base + lane] = e / sum;

    float bvv = mval;
    int   bi  = lane;
#pragma unroll
    for (int off = 16; off > 0; off >>= 1) {
      float ov = __shfl_xor(bvv, off, 32);
      int   oi = __shfl_xor(bi, off, 32);
      if (ov > bvv || (ov == bvv && oi < bi)) { bvv = ov; bi = oi; }
    }
    flag = flag || (lane == bi);
  }
}

// ---------------------------------------------------------------------------
extern "C" void kernel_launch(void* const* d_in, const int* in_sizes, int n_in,
                              void* d_out, int out_size, void* d_ws,
                              size_t ws_size, hipStream_t stream) {
  const float* input_hidden = (const float*)d_in[0];
  const float* tgt          = (const float*)d_in[1];
  const int*   pos_label    = (const int*)d_in[2];
  // d_in[3] = max_length (scalar, == T, unused)
  const float* S    = (const float*)d_in[4];
  const float* W_ih = (const float*)d_in[5];
  const float* W_hh = (const float*)d_in[6];
  const float* b_ih = (const float*)d_in[7];
  const float* b_hh = (const float*)d_in[8];
  const float* Wl_w = (const float*)d_in[9];
  const float* Wl_b = (const float*)d_in[10];
  float* out = (float*)d_out;

  char* w = (char*)d_ws;
  auto carve = [&](size_t bytes) -> void* {
    void* p = (void*)w;
    w += (bytes + 255) & ~(size_t)255;
    return p;
  };
  __bf16* ih_bf  = (__bf16*)carve((size_t)B * L * H * 2);
  __bf16* tgt_bf = (__bf16*)carve((size_t)B * T * H * 2);
  __bf16* Wih_bf = (__bf16*)carve((size_t)H3 * H * 2);
  __bf16* Whh_bf = (__bf16*)carve((size_t)H3 * H * 2);
  __bf16* Wl_bf  = (__bf16*)carve((size_t)H * H * 2);
  float*  gi0    = (float*)carve((size_t)H3 * 4);
  float*  GI     = (float*)carve((size_t)B * T * H3 * 4);
  float*  hA     = (float*)carve((size_t)B * H * 4);
  float*  hB     = (float*)carve((size_t)B * H * 4);
  __bf16* hA_bf  = (__bf16*)carve((size_t)B * H * 2);
  __bf16* hB_bf  = (__bf16*)carve((size_t)B * H * 2);

  // ---- prep (plain casts; weights keep native [N,K] layout) --------------
  {
    int n = B * L * H;
    cast_f32_to_bf16<<<(n + 255) / 256, 256, 0, stream>>>(input_hidden, ih_bf, n);
    n = B * T * H;
    cast_f32_to_bf16<<<(n + 255) / 256, 256, 0, stream>>>(tgt, tgt_bf, n);
    n = H3 * H;
    cast_f32_to_bf16<<<(n + 255) / 256, 256, 0, stream>>>(W_ih, Wih_bf, n);
    cast_f32_to_bf16<<<(n + 255) / 256, 256, 0, stream>>>(W_hh, Whh_bf, n);
    n = H * H;
    cast_f32_to_bf16<<<(n + 255) / 256, 256, 0, stream>>>(Wl_w, Wl_bf, n);
    gi0_kernel<<<(H3 + 255) / 256, 256, 0, stream>>>(S, W_ih, b_ih, gi0);
    n = B * H;
    mean_kernel<<<(n + 255) / 256, 256, 0, stream>>>(input_hidden, hA, hA_bf);
  }

  // ---- big parallel GEMM: GI[b*T+t] = tgt[b,t] @ W_ih^T + b_ih ----------
  {
    const int M = B * T;                       // 15360
    gemm_bias_kernel<<<dim3(M / 64, H3 / 128), 256, 0, stream>>>(
        tgt_bf, Wih_bf, b_ih, GI, M, H3, H);
  }

  // ---- serial GRU + attention chain -------------------------------------
  for (int t = 0; t < T; ++t) {
    const float*  h_cur    = (t & 1) ? hB : hA;
    const __bf16* h_cur_bf = (t & 1) ? hB_bf : hA_bf;
    float*        h_nxt    = (t & 1) ? hA : hB;
    __bf16*       h_nxt_bf = (t & 1) ? hA_bf : hB_bf;

    step_h_kernel<<<dim3(B / 64, H / 64), 256, 0, stream>>>(
        h_cur, h_cur_bf, h_nxt, h_nxt_bf, Whh_bf, b_hh, GI, gi0,
        t - 1, (t == 0) ? 1 : 0);

    step_attn_kernel<<<B / 16, 256, 0, stream>>>(
        h_nxt_bf, Wl_bf, Wl_b, ih_bf, out, t);
  }

  // ---- softmax / argmax / flag chain ------------------------------------
  plan_kernel<<<B / 8, 256, 0, stream>>>(pos_label, out);

  (void)in_sizes; (void)n_in; (void)out_size; (void)ws_size;
}
